// GraphTripleConv_455266533448
// MI455X (gfx1250) — matrix operand order repaired and checked by reading, not compiled, not run on hardware
//
#include <hip/hip_runtime.h>
#include <hip/hip_bf16.h>

// Problem sizes (fixed by the reference)
#define O_N    50000
#define T_N    100000
#define DIN_N  128
#define H_DIM  512
#define DOUT_N 128

typedef __attribute__((ext_vector_type(16))) __bf16 v16bf;
typedef __attribute__((ext_vector_type(8)))  float  v8f;

union Frag16 { v16bf v; unsigned u[8]; };

__device__ __forceinline__ v8f zero_v8f() {
  v8f z = {0.f, 0.f, 0.f, 0.f, 0.f, 0.f, 0.f, 0.f};
  return z;
}

// D = A*B + C, bf16 operands, f32 accumulate (v_wmma_f32_16x16x32_bf16)
__device__ __forceinline__ v8f wmma_bf16(v16bf a, v16bf b, v8f c) {
  return __builtin_amdgcn_wmma_f32_16x16x32_bf16(
      /*neg_a=*/false, a, /*neg_b=*/false, b,
      /*c_mod=*/(short)0, c, /*reuse_a=*/false, /*reuse_b=*/false);
}

// ---- CDNA5 async global->LDS copy (ASYNCcnt-tracked), per-lane 16 bytes.
// LDS address operand = low 32 bits of the flat shared pointer (ISA 10.2:
// flat LDS addresses keep the group-segment offset in addr[31:0]).
__device__ __forceinline__ void async_copy_b128(const void* gsrc, void* ldst) {
  unsigned lds_addr = (unsigned)(size_t)ldst;
  asm volatile("global_load_async_to_lds_b128 %0, %1, off"
               :
               : "v"(lds_addr), "v"(gsrc)
               : "memory");
}

__device__ __forceinline__ void wait_asynccnt0() {
  asm volatile("s_wait_asynccnt 0" ::: "memory");
}

// A fragment: 16x32 bf16 tile of a row-major matrix (compile-time stride).
// Layout per ISA 7.12.2 (16-bit A 16x32): lane%16 = M; lanes 0-15 hold
// K={0..7,16..23}, lanes 16-31 hold K={8..15,24..31}, 2 K-values per VGPR.
template <int STRIDE>
__device__ __forceinline__ v16bf load_a_frag(const __bf16* base, int m0, int k0, int lane) {
  Frag16 f;
  const __bf16* row = base + (m0 + (lane & 15)) * STRIDE + k0;
  const int h8 = (lane & 16) >> 1;  // 0 or 8
#pragma unroll
  for (int j = 0; j < 8; ++j) {
    const int k = (j < 4) ? (2 * j + h8) : (16 + 2 * (j - 4) + h8);
    f.u[j] = *reinterpret_cast<const unsigned*>(row + k);  // pair (k, k+1)
  }
  return f.v;
}

// B fragment: 32x16 bf16 tile of W^T, where W is (N x K) row-major (weights).
// Layout: lane%16 = N; lanes 0-15 hold K={0..7,16..23}, lanes 16-31 hold K={8..15,24..31}.
__device__ __forceinline__ v16bf load_b_frag(const __bf16* W, int K, int n0, int k0, int lane) {
  Frag16 f;
  const __bf16* row = W + (size_t)(n0 + (lane & 15)) * K + (k0 + ((lane & 16) >> 1));
#pragma unroll
  for (int j = 0; j < 8; ++j)
    f.u[j] = *reinterpret_cast<const unsigned*>(row + ((j < 4) ? 2 * j : 16 + 2 * (j - 4)));
  return f.v;
}

// ---------------------------------------------------------------- utilities

__global__ void k_f32_to_bf16(const float* __restrict__ src, __bf16* __restrict__ dst, long long n) {
  long long i = (long long)blockIdx.x * blockDim.x + threadIdx.x;
  const long long stride = (long long)gridDim.x * blockDim.x;
  for (; i < n; i += stride) dst[i] = (__bf16)src[i];
}

__global__ void k_zero_f32(float* __restrict__ p, long long n) {
  long long i = (long long)blockIdx.x * blockDim.x + threadIdx.x;
  const long long stride = (long long)gridDim.x * blockDim.x;
  for (; i < n; i += stride) p[i] = 0.0f;
}

// ------------------------------------------------------- fused triple MLP
// One block = 64 triples. Gather -> GEMM1(relu) -> GEMM2(relu) -> scatter.
// LDS: cur_t 64x384 bf16 (48KB) + h1 64x512 bf16 (64KB) + 2x64 int idx.
#define SMEM_TRI (64 * 384 * 2 + 64 * 512 * 2 + 2 * 64 * 4)

__global__ void __launch_bounds__(256)
k_triples(const __bf16* __restrict__ objs, const __bf16* __restrict__ preds,
          const int* __restrict__ edges,
          const __bf16* __restrict__ w1a, const float* __restrict__ b1a,
          const __bf16* __restrict__ w1b, const float* __restrict__ b1b,
          float* __restrict__ pooled, float* __restrict__ counts,
          float* __restrict__ out_p) {
  extern __shared__ char smem[];
  __bf16* curt = (__bf16*)smem;                              // 64 x 384
  __bf16* h1   = (__bf16*)(smem + 64 * 384 * 2);             // 64 x 512
  int* sidx    = (int*)(smem + 64 * 384 * 2 + 64 * 512 * 2); // 64
  int* oidx    = sidx + 64;                                  // 64

  const int t0   = blockIdx.x * 64;
  const int tid  = threadIdx.x;
  const int lane = tid & 31;
  const int wv   = tid >> 5;  // wave id 0..7

  // Pull W1b toward L2 early (it dominates stage-2 reads).
  __builtin_prefetch(w1b + (size_t)tid * 2048, 0, 1);

  if (tid < 64) {
    const int t = t0 + tid;
    int s = 0, o = 0;
    if (t < T_N) {
      s = edges[2 * t];
      o = edges[2 * t + 1];
      atomicAdd(&counts[s], 1.0f);
      atomicAdd(&counts[o], 1.0f);
    }
    sidx[tid] = s;
    oidx[tid] = o;
  }
  __syncthreads();

  // Gather cur_t tile straight into LDS with async b128 copies:
  // 64 rows x 48 chunks of 16B (row = [obj[s] | pred | obj[o]], 768B).
  for (int i = tid; i < 64 * 48; i += 256) {
    const int r = i / 48;
    const int c = i - r * 48;  // 16B chunk within row
    const int t = t0 + r;
    char* dst = (char*)curt + (size_t)i * 16;  // row-major: r*768 + c*16
    if (t < T_N) {
      const char* g;
      if (c < 16)
        g = (const char*)(objs + (size_t)sidx[r] * DIN_N) + (size_t)c * 16;
      else if (c < 32)
        g = (const char*)(preds + (size_t)t * DIN_N) + (size_t)(c - 16) * 16;
      else
        g = (const char*)(objs + (size_t)oidx[r] * DIN_N) + (size_t)(c - 32) * 16;
      async_copy_b128(g, dst);
    } else {
      unsigned* d = (unsigned*)dst;
      d[0] = 0u; d[1] = 0u; d[2] = 0u; d[3] = 0u;
    }
  }
  wait_asynccnt0();
  __syncthreads();

  // ---- GEMM1: h1 = relu(cur_t @ W1a^T + b1a).  M=64, N=512, K=384.
  for (int nt = wv; nt < 32; nt += 8) {
    v8f acc[4];
#pragma unroll
    for (int m = 0; m < 4; ++m) acc[m] = zero_v8f();
    for (int k0 = 0; k0 < 384; k0 += 32) {
      const v16bf bf = load_b_frag(w1a, 384, nt * 16, k0, lane);
#pragma unroll
      for (int m = 0; m < 4; ++m) {
        const v16bf af = load_a_frag<384>(curt, m * 16, k0, lane);
        acc[m] = wmma_bf16(af, bf, acc[m]);
      }
    }
    const int col    = nt * 16 + (lane & 15);
    const float bias = b1a[col];
    const int rbase  = (lane & 16) >> 1;  // C layout: +8 rows for upper half-wave
#pragma unroll
    for (int m = 0; m < 4; ++m)
#pragma unroll
      for (int j = 0; j < 8; ++j) {
        const int row = m * 16 + rbase + j;
        h1[row * H_DIM + col] = (__bf16)fmaxf(acc[m][j] + bias, 0.0f);
      }
  }
  __syncthreads();

  // ---- GEMM2 + split/scatter: new_t = relu(h1 @ W1b^T + b1b).  N=1152, K=512.
  for (int nt = wv; nt < 72; nt += 8) {
    v8f acc[4];
#pragma unroll
    for (int m = 0; m < 4; ++m) acc[m] = zero_v8f();
    for (int k0 = 0; k0 < 512; k0 += 32) {
      const v16bf bf = load_b_frag(w1b, 512, nt * 16, k0, lane);
#pragma unroll
      for (int m = 0; m < 4; ++m) {
        const v16bf af = load_a_frag<H_DIM>(h1, m * 16, k0, lane);
        acc[m] = wmma_bf16(af, bf, acc[m]);
      }
    }
    const int col    = nt * 16 + (lane & 15);
    const float bias = b1b[col];
    const int rbase  = (lane & 16) >> 1;
#pragma unroll
    for (int m = 0; m < 4; ++m)
#pragma unroll
      for (int j = 0; j < 8; ++j) {
        const int row = m * 16 + rbase + j;
        const int t   = t0 + row;
        if (t < T_N) {
          const float v = fmaxf(acc[m][j] + bias, 0.0f);
          if (col < H_DIM)
            atomicAdd(&pooled[(size_t)sidx[row] * H_DIM + col], v);            // new_s
          else if (col < H_DIM + DOUT_N)
            out_p[(size_t)t * DOUT_N + (col - H_DIM)] = v;                      // new_p
          else
            atomicAdd(&pooled[(size_t)oidx[row] * H_DIM + (col - H_DIM - DOUT_N)], v); // new_o
        }
      }
  }
}

// ------------------------------------------------------- fused object MLP
// One block = 64 objects. normalize -> GEMM3(relu) -> GEMM4(relu).
// LDS: pooled 64x512 bf16 (64KB) + h2 64x512 bf16 (64KB) + 64 f32 scales.
#define SMEM_OBJ (64 * 512 * 2 * 2 + 64 * 4)

__global__ void __launch_bounds__(256)
k_obj(const float* __restrict__ pooled, const float* __restrict__ counts,
      const __bf16* __restrict__ w2a, const float* __restrict__ b2a,
      const __bf16* __restrict__ w2b, const float* __restrict__ b2b,
      float* __restrict__ out_obj) {
  extern __shared__ char smem[];
  __bf16* pb  = (__bf16*)smem;                       // 64 x 512
  __bf16* h2  = (__bf16*)(smem + 64 * 512 * 2);      // 64 x 512
  float*  scl = (float*)(smem + 2 * 64 * 512 * 2);   // 64

  const int o0   = blockIdx.x * 64;
  const int tid  = threadIdx.x;
  const int lane = tid & 31;
  const int wv   = tid >> 5;

  if (tid < 64) {
    const int ob  = o0 + tid;
    const float c = (ob < O_N) ? counts[ob] : 1.0f;
    scl[tid] = 1.0f / fmaxf(c, 1.0f);
  }
  __syncthreads();

  // Normalize pooled by degree and stage as bf16 (b128 loads, packed stores).
  for (int i = tid; i < 64 * 128; i += 256) {
    const int r  = i >> 7;
    const int c4 = (i & 127) << 2;
    const int ob = o0 + r;
    float4 v = make_float4(0.f, 0.f, 0.f, 0.f);
    if (ob < O_N) v = *reinterpret_cast<const float4*>(pooled + (size_t)ob * H_DIM + c4);
    const float s = scl[r];
    __bf16* d = pb + r * H_DIM + c4;
    d[0] = (__bf16)(v.x * s);
    d[1] = (__bf16)(v.y * s);
    d[2] = (__bf16)(v.z * s);
    d[3] = (__bf16)(v.w * s);
  }
  __syncthreads();

  // ---- GEMM3: h2 = relu(pooled @ W2a^T + b2a).  N=512, K=512.
  for (int nt = wv; nt < 32; nt += 8) {
    v8f acc[4];
#pragma unroll
    for (int m = 0; m < 4; ++m) acc[m] = zero_v8f();
    for (int k0 = 0; k0 < 512; k0 += 32) {
      const v16bf bf = load_b_frag(w2a, 512, nt * 16, k0, lane);
#pragma unroll
      for (int m = 0; m < 4; ++m) {
        const v16bf af = load_a_frag<H_DIM>(pb, m * 16, k0, lane);
        acc[m] = wmma_bf16(af, bf, acc[m]);
      }
    }
    const int col    = nt * 16 + (lane & 15);
    const float bias = b2a[col];
    const int rbase  = (lane & 16) >> 1;
#pragma unroll
    for (int m = 0; m < 4; ++m)
#pragma unroll
      for (int j = 0; j < 8; ++j) {
        const int row = m * 16 + rbase + j;
        h2[row * H_DIM + col] = (__bf16)fmaxf(acc[m][j] + bias, 0.0f);
      }
  }
  __syncthreads();

  // ---- GEMM4: new_obj = relu(h2 @ W2b^T + b2b).  N=128, K=512.
  {
    const int nt = wv;  // 8 waves cover 8 N-tiles of 16
    v8f acc[4];
#pragma unroll
    for (int m = 0; m < 4; ++m) acc[m] = zero_v8f();
    for (int k0 = 0; k0 < 512; k0 += 32) {
      const v16bf bf = load_b_frag(w2b, 512, nt * 16, k0, lane);
#pragma unroll
      for (int m = 0; m < 4; ++m) {
        const v16bf af = load_a_frag<H_DIM>(h2, m * 16, k0, lane);
        acc[m] = wmma_bf16(af, bf, acc[m]);
      }
    }
    const int col    = nt * 16 + (lane & 15);
    const float bias = b2b[col];
    const int rbase  = (lane & 16) >> 1;
#pragma unroll
    for (int m = 0; m < 4; ++m)
#pragma unroll
      for (int j = 0; j < 8; ++j) {
        const int row = m * 16 + rbase + j;
        const int ob  = o0 + row;
        if (ob < O_N)
          out_obj[(size_t)ob * DOUT_N + col] = fmaxf(acc[m][j] + bias, 0.0f);
      }
  }
}

// ----------------------------------------------------------------- launch

extern "C" void kernel_launch(void* const* d_in, const int* in_sizes, int n_in,
                              void* d_out, int out_size, void* d_ws, size_t ws_size,
                              hipStream_t stream) {
  (void)in_sizes; (void)n_in; (void)out_size; (void)ws_size;

  const float* obj_vecs  = (const float*)d_in[0];
  const float* pred_vecs = (const float*)d_in[1];
  const int*   edges     = (const int*)d_in[2];
  const float* W1a = (const float*)d_in[3];
  const float* b1a = (const float*)d_in[4];
  const float* W1b = (const float*)d_in[5];
  const float* b1b = (const float*)d_in[6];
  const float* W2a = (const float*)d_in[7];
  const float* b2a = (const float*)d_in[8];
  const float* W2b = (const float*)d_in[9];
  const float* b2b = (const float*)d_in[10];

  float* out_obj = (float*)d_out;                         // (O, 128)
  float* out_p   = out_obj + (size_t)O_N * DOUT_N;        // (T, 128)

  // Workspace carve-out (256B aligned slabs).
  char* ws = (char*)d_ws;
  size_t off = 0;
  auto carve = [&](size_t bytes) -> void* {
    void* p = ws + off;
    off = (off + bytes + 255) & ~(size_t)255;
    return p;
  };
  __bf16* objs_bf  = (__bf16*)carve((size_t)O_N * DIN_N * 2);
  __bf16* preds_bf = (__bf16*)carve((size_t)T_N * DIN_N * 2);
  __bf16* w1a_bf   = (__bf16*)carve((size_t)H_DIM * (3 * DIN_N) * 2);
  __bf16* w1b_bf   = (__bf16*)carve((size_t)(2 * H_DIM + DOUT_N) * H_DIM * 2);
  __bf16* w2a_bf   = (__bf16*)carve((size_t)H_DIM * H_DIM * 2);
  __bf16* w2b_bf   = (__bf16*)carve((size_t)DOUT_N * H_DIM * 2);
  float*  pooled   = (float*)carve((size_t)O_N * H_DIM * 4);
  float*  counts   = (float*)carve((size_t)O_N * 4);

  // Precision staging: f32 -> bf16 (WMMA operand format), f32 accumulate later.
  k_f32_to_bf16<<<2048, 256, 0, stream>>>(obj_vecs,  objs_bf,  (long long)O_N * DIN_N);
  k_f32_to_bf16<<<2048, 256, 0, stream>>>(pred_vecs, preds_bf, (long long)T_N * DIN_N);
  k_f32_to_bf16<<<256,  256, 0, stream>>>(W1a, w1a_bf, (long long)H_DIM * (3 * DIN_N));
  k_f32_to_bf16<<<256,  256, 0, stream>>>(W1b, w1b_bf, (long long)(2 * H_DIM + DOUT_N) * H_DIM);
  k_f32_to_bf16<<<256,  256, 0, stream>>>(W2a, w2a_bf, (long long)H_DIM * H_DIM);
  k_f32_to_bf16<<<256,  256, 0, stream>>>(W2b, w2b_bf, (long long)DOUT_N * H_DIM);

  // Scatter targets must start at zero every call.
  k_zero_f32<<<2048, 256, 0, stream>>>(pooled, (long long)O_N * H_DIM);
  k_zero_f32<<<128,  256, 0, stream>>>(counts, (long long)O_N);

  // >64KB dynamic LDS (CDNA5 WGP has 320KB).
  hipFuncSetAttribute((const void*)k_triples, hipFuncAttributeMaxDynamicSharedMemorySize, SMEM_TRI);
  hipFuncSetAttribute((const void*)k_obj,     hipFuncAttributeMaxDynamicSharedMemorySize, SMEM_OBJ);

  k_triples<<<(T_N + 63) / 64, 256, SMEM_TRI, stream>>>(
      objs_bf, preds_bf, edges, w1a_bf, b1a, w1b_bf, b1b, pooled, counts, out_p);

  k_obj<<<(O_N + 63) / 64, 256, SMEM_OBJ, stream>>>(
      pooled, counts, w2a_bf, b2a, w2b_bf, b2b, out_obj);
}
